// MaskedAttentionHead_41729902248434
// MI455X (gfx1250) — matrix operand order repaired and checked by reading, not compiled, not run on hardware
//
#include <hip/hip_runtime.h>
#include <hip/hip_bf16.h>

typedef __attribute__((ext_vector_type(16))) _Float16 v16h;
typedef __attribute__((ext_vector_type(8)))  _Float16 v8h;
typedef __attribute__((ext_vector_type(8)))  float    v8f;
typedef __attribute__((ext_vector_type(4)))  float    v4f;

#define BATCH 4
#define SEQ   4096
#define EMB   1024
#define HD    64
#define ROWS  (BATCH * SEQ)

static __device__ __forceinline__ v16h cat8(v8h a, v8h b) {
  return __builtin_shufflevector(a, b, 0,1,2,3,4,5,6,7,8,9,10,11,12,13,14,15);
}

// ---------------------------------------------------------------------------
// Kernel 0: convert Wq/Wk/Wv (fp32 [1024][64]) -> f16 transposed wT[3][64][1024]
// so that projection B-fragments are pairs of contiguous 16B loads.
// ---------------------------------------------------------------------------
__global__ void __launch_bounds__(256) mha_wconv(const float* __restrict__ Wq,
                                                 const float* __restrict__ Wk,
                                                 const float* __restrict__ Wv,
                                                 _Float16* __restrict__ wT) {
  int idx = blockIdx.x * 256 + threadIdx.x;      // t*65536 + d*1024 + e
  if (idx >= 3 * HD * EMB) return;
  int t = idx >> 16;
  int d = (idx >> 10) & 63;
  int e = idx & 1023;
  const float* W = (t == 0) ? Wq : ((t == 1) ? Wk : Wv);
  wT[idx] = (_Float16)W[e * HD + d];
}

// ---------------------------------------------------------------------------
// Kernel 1: projections. One wave computes a 16-row x 64-col tile of Q, K and
// V simultaneously (H loaded once). A-tile is software-pipelined; B-fragments
// load as named groups of 4 so waits amortize over 4 WMMAs.
// Q,K stored f16 row-major [16384][64]; V stored transposed vT[b][64][4096].
// ---------------------------------------------------------------------------
__global__ void __launch_bounds__(128) mha_proj(const float* __restrict__ H,
                                                const _Float16* __restrict__ wT,
                                                _Float16* __restrict__ qf,
                                                _Float16* __restrict__ kf,
                                                _Float16* __restrict__ vT) {
  const int wave = threadIdx.x >> 5, lane = threadIdx.x & 31;
  const int m = lane & 15, hi = lane >> 4;
  const int tile = blockIdx.x * 4 + wave;
  const int row0 = tile * 16;
  const float* hrow = H + (size_t)(row0 + m) * EMB;

  v8f zero = {};
  v8f acc[3][4];
#pragma unroll
  for (int t = 0; t < 3; ++t)
#pragma unroll
    for (int n = 0; n < 4; ++n) acc[t][n] = zero;

  // Prologue: first A tile (k0 = 0)
  v4f x0 = *(const v4f*)(hrow + 8 * hi);
  v4f x1 = *(const v4f*)(hrow + 8 * hi + 4);
  v4f x2 = *(const v4f*)(hrow + 16 + 8 * hi);
  v4f x3 = *(const v4f*)(hrow + 16 + 8 * hi + 4);

  for (int k0 = 0; k0 < EMB; k0 += 32) {
    // Convert staged A tile. Layout: j<8 -> K=8*hi+j ; j>=8 -> K=16+8*hi+(j-8)
    v16h a;
#pragma unroll
    for (int j = 0; j < 4; ++j) {
      a[j]      = (_Float16)x0[j];
      a[4 + j]  = (_Float16)x1[j];
      a[8 + j]  = (_Float16)x2[j];
      a[12 + j] = (_Float16)x3[j];
    }
    // Pipeline: issue next A-tile loads now; waits resolve at next iteration.
    if (k0 + 32 < EMB) {
      x0 = *(const v4f*)(hrow + k0 + 32 + 8 * hi);
      x1 = *(const v4f*)(hrow + k0 + 32 + 8 * hi + 4);
      x2 = *(const v4f*)(hrow + k0 + 48 + 8 * hi);
      x3 = *(const v4f*)(hrow + k0 + 48 + 8 * hi + 4);
    }
#pragma unroll
    for (int t = 0; t < 3; ++t) {
      const _Float16* wp = wT + (size_t)t * HD * EMB + (size_t)m * EMB + k0 + 8 * hi;
      // B fragments: lane col = 16n+m, j<8 -> K = 8*hi + j ; j>=8 -> K = 16+8*hi+(j-8)
      v16h bf0 = cat8(*(const v8h*)(wp), *(const v8h*)(wp + 16));
      v16h bf1 = cat8(*(const v8h*)(wp + (size_t)16 * EMB), *(const v8h*)(wp + (size_t)16 * EMB + 16));
      v16h bf2 = cat8(*(const v8h*)(wp + (size_t)32 * EMB), *(const v8h*)(wp + (size_t)32 * EMB + 16));
      v16h bf3 = cat8(*(const v8h*)(wp + (size_t)48 * EMB), *(const v8h*)(wp + (size_t)48 * EMB + 16));
      acc[t][0] = __builtin_amdgcn_wmma_f32_16x16x32_f16(false, a, false, bf0,
                                                         (short)0, acc[t][0], false, false);
      acc[t][1] = __builtin_amdgcn_wmma_f32_16x16x32_f16(false, a, false, bf1,
                                                         (short)0, acc[t][1], false, false);
      acc[t][2] = __builtin_amdgcn_wmma_f32_16x16x32_f16(false, a, false, bf2,
                                                         (short)0, acc[t][2], false, false);
      acc[t][3] = __builtin_amdgcn_wmma_f32_16x16x32_f16(false, a, false, bf3,
                                                         (short)0, acc[t][3], false, false);
    }
  }

  const int b  = row0 >> 12;        // row0 / 4096
  const int rb = row0 & (SEQ - 1);  // row within batch
#pragma unroll
  for (int n = 0; n < 4; ++n) {
    const int col = 16 * n + m;
#pragma unroll
    for (int r = 0; r < 8; ++r) {
      const size_t row = (size_t)(row0 + r + 8 * hi);
      qf[row * HD + col] = (_Float16)acc[0][n][r];
      kf[row * HD + col] = (_Float16)acc[1][n][r];
    }
    v8h pv;
#pragma unroll
    for (int r = 0; r < 8; ++r) pv[r] = (_Float16)acc[2][n][r];
    *(v8h*)(vT + ((size_t)b * HD + col) * SEQ + rb + 8 * hi) = pv;
  }
}

// ---------------------------------------------------------------------------
// Kernel 2: causal flash attention. One wave owns 16 query rows; iterates over
// key blocks of 32. V-fragment loads are issued BEFORE the softmax VALU block
// so ~150 VALU ops hide their latency (the LDS fence would otherwise block
// the compiler from hoisting them).
// ---------------------------------------------------------------------------
__global__ void __launch_bounds__(128) mha_attn(const _Float16* __restrict__ qf,
                                                const _Float16* __restrict__ kf,
                                                const _Float16* __restrict__ vT,
                                                float* __restrict__ out) {
  __shared__ __align__(16) _Float16 plds[4][16 * 32];
  const int wave = threadIdx.x >> 5, lane = threadIdx.x & 31;
  const int m = lane & 15, hi = lane >> 4;
  const int tile = blockIdx.x * 4 + wave;
  const int b  = tile >> 8;               // 256 q-tiles per batch
  const int q0 = (tile & 255) * 16;       // query row within batch
  _Float16* pl = plds[wave];

  // Q A-fragments (16x32 each, head dim split in two)
  const _Float16* qrow = qf + ((size_t)b * SEQ + q0) * HD + (size_t)m * HD;
  v16h qa[2];
#pragma unroll
  for (int f = 0; f < 2; ++f) {
    v8h a0 = *(const v8h*)(qrow + f * 32 + 16 * hi);
    v8h a1 = *(const v8h*)(qrow + f * 32 + 16 * hi + 8);
    qa[f] = cat8(a0, a1);
  }

  v8f zero = {};
  v8f o[4];
#pragma unroll
  for (int t = 0; t < 4; ++t) o[t] = zero;
  float mr[8], lr[8];
#pragma unroll
  for (int r = 0; r < 8; ++r) { mr[r] = -1.0e30f; lr[r] = 0.0f; }

  const float SL2E = 0.125f * 1.44269504088896340736f;  // scale * log2(e)
  const int nkb = (q0 + 16 + 31) >> 5;                  // causal: keys 0 .. q0+15
  const _Float16* kbase = kf + (size_t)b * SEQ * HD;
  const _Float16* vbase = vT + (size_t)b * HD * SEQ;

  for (int ib = 0; ib < nkb; ++ib) {
    const int kb = ib * 32;
    const _Float16* kp = kbase + (size_t)kb * HD;
    if (ib + 1 < nkb) {
      __builtin_prefetch(kp + 32 * HD, 0, 2);
      __builtin_prefetch(vbase + (size_t)m * SEQ + kb + 32, 0, 2);
    }

    // K B-fragments, named group (waits amortize over the 4 S-WMMAs).
    // tile0: rows kb+m ; tile1: rows kb+16+m ; halves j -> K = 16*hi + j
    v16h kf00 = *(const v16h*)(kp + (size_t)m * HD + 16 * hi);            // f=0 tile0
    v16h kf01 = *(const v16h*)(kp + (size_t)(16 + m) * HD + 16 * hi);     // f=0 tile1
    v16h kf10 = *(const v16h*)(kp + (size_t)m * HD + 32 + 16 * hi);       // f=1 tile0
    v16h kf11 = *(const v16h*)(kp + (size_t)(16 + m) * HD + 32 + 16 * hi);// f=1 tile1

    // V B-fragments issued NOW so the softmax VALU block hides their latency.
    v16h vb[4];
#pragma unroll
    for (int t = 0; t < 4; ++t)
      vb[t] = *(const v16h*)(vbase + (size_t)(16 * t + m) * SEQ + kb + 16 * hi);

    // S tiles: s0 = Q . K[kb..kb+15]^T ; s1 = Q . K[kb+16..kb+31]^T
    v8f s0 = zero, s1 = zero;
    s0 = __builtin_amdgcn_wmma_f32_16x16x32_f16(false, qa[0], false, kf00,
                                                (short)0, s0, false, false);
    s0 = __builtin_amdgcn_wmma_f32_16x16x32_f16(false, qa[1], false, kf10,
                                                (short)0, s0, false, false);
    s1 = __builtin_amdgcn_wmma_f32_16x16x32_f16(false, qa[0], false, kf01,
                                                (short)0, s1, false, false);
    s1 = __builtin_amdgcn_wmma_f32_16x16x32_f16(false, qa[1], false, kf11,
                                                (short)0, s1, false, false);

    // online softmax over the 32 key columns (rows live across 16 lanes)
#pragma unroll
    for (int r = 0; r < 8; ++r) {
      const int qi = q0 + r + 8 * hi;
      float x0 = (kb + m      <= qi) ? s0[r] : -3.0e38f;
      float x1 = (kb + 16 + m <= qi) ? s1[r] : -3.0e38f;
      float mx = fmaxf(x0, x1);
      mx = fmaxf(mx, __shfl_xor(mx, 1));
      mx = fmaxf(mx, __shfl_xor(mx, 2));
      mx = fmaxf(mx, __shfl_xor(mx, 4));
      mx = fmaxf(mx, __shfl_xor(mx, 8));
      const float mnew = fmaxf(mr[r], mx);
      const float corr = __builtin_exp2f((mr[r] - mnew) * SL2E);
      const float p0 = __builtin_exp2f((x0 - mnew) * SL2E);
      const float p1 = __builtin_exp2f((x1 - mnew) * SL2E);
      float ps = p0 + p1;
      ps += __shfl_xor(ps, 1);
      ps += __shfl_xor(ps, 2);
      ps += __shfl_xor(ps, 4);
      ps += __shfl_xor(ps, 8);
      lr[r] = lr[r] * corr + ps;
      mr[r] = mnew;
#pragma unroll
      for (int t = 0; t < 4; ++t) o[t][r] *= corr;
      // stage P (f16) in LDS: row r+8*hi, cols m and 16+m
      pl[(r + 8 * hi) * 32 + m]      = (_Float16)p0;
      pl[(r + 8 * hi) * 32 + 16 + m] = (_Float16)p1;
    }

    // LDS RAW fence: all lanes' P stores visible before A-fragment reads
    asm volatile("s_wait_dscnt 0" ::: "memory");

    v8h g0 = *(const v8h*)(pl + m * 32 + 16 * hi);
    v8h g1 = *(const v8h*)(pl + m * 32 + 16 * hi + 8);
    v16h pa = cat8(g0, g1);

    // O += P(16x32) * V(32x64); V fragments already resident
#pragma unroll
    for (int t = 0; t < 4; ++t)
      o[t] = __builtin_amdgcn_wmma_f32_16x16x32_f16(false, pa, false, vb[t],
                                                    (short)0, o[t], false, false);
  }

  float* op = out + ((size_t)b * SEQ + q0) * HD;
#pragma unroll
  for (int r = 0; r < 8; ++r) {
    const float inv = 1.0f / lr[r];
#pragma unroll
    for (int t = 0; t < 4; ++t)
      op[(size_t)(r + 8 * hi) * HD + 16 * t + m] = o[t][r] * inv;
  }
}

// ---------------------------------------------------------------------------
extern "C" void kernel_launch(void* const* d_in, const int* in_sizes, int n_in,
                              void* d_out, int out_size, void* d_ws, size_t ws_size,
                              hipStream_t stream) {
  (void)in_sizes; (void)n_in; (void)out_size; (void)ws_size;
  const float* H  = (const float*)d_in[0];
  const float* Wq = (const float*)d_in[1];
  const float* Wk = (const float*)d_in[2];
  const float* Wv = (const float*)d_in[3];
  float* out = (float*)d_out;

  // Workspace layout (f16 elements):
  _Float16* qf = (_Float16*)d_ws;                  // [16384][64]
  _Float16* kf = qf + (size_t)ROWS * HD;           // [16384][64]
  _Float16* vT = kf + (size_t)ROWS * HD;           // [4][64][4096]
  _Float16* wT = vT + (size_t)BATCH * HD * SEQ;    // [3][64][1024]

  mha_wconv<<<(3 * HD * EMB + 255) / 256, 256, 0, stream>>>(Wq, Wk, Wv, wT);
  mha_proj<<<ROWS / 16 / 4, 128, 0, stream>>>(H, wT, qf, kf, vT);
  mha_attn<<<(BATCH * SEQ / 16) / 4, 128, 0, stream>>>(qf, kf, vT, out);
}
